// LinearAttention_62380105007289
// MI455X (gfx1250) — matrix operand order repaired and checked by reading, not compiled, not run on hardware
//
#include <hip/hip_runtime.h>
#include <hip/hip_bf16.h>
#include <math.h>

typedef __attribute__((ext_vector_type(16))) _Float16 v16h;
typedef __attribute__((ext_vector_type(8)))  _Float16 v8h;
typedef __attribute__((ext_vector_type(8)))  float    v8f;
typedef __attribute__((ext_vector_type(4)))  float    f32x4;
typedef __attribute__((ext_vector_type(4)))  unsigned u32x4;
typedef __attribute__((ext_vector_type(4)))  int      i32x4;
typedef __attribute__((ext_vector_type(8)))  int      i32x8;

#define DIM   1024
#define SEQ   4096
#define BATCH 4
#define HEADS 16
#define HD    64
#define MTOT  (BATCH*SEQ)   /* 16384 rows */

/* LDS row stride in halves: 32 data + 16 pad (=32B) keeps v16h/v8h loads aligned
   and matches the TDM pad descriptor (16 DWORDs data, 8 DWORDs pad per row). */
#define LDST 48
#define TILEHALVES (128 * LDST)

/* ----------------------- TDM availability probes -------------------------- */
#if defined(__has_builtin)
#  if __has_builtin(__builtin_amdgcn_tensor_load_to_lds) && \
      __has_builtin(__builtin_amdgcn_s_wait_tensorcnt)
#    define HAVE_TDM 1
#  endif
#endif
#ifndef HAVE_TDM
#  define HAVE_TDM 0
#endif

#if defined(__has_include)
#  if __has_include(<hip/amd_detail/amd_gfx1250_TDM.h>)
#    define TDM_6ARG 1
#  else
#    define TDM_6ARG 0
#  endif
#else
#  define TDM_6ARG 0
#endif

/* ---------------- WMMA helpers (CDNA5 16x16x32 f16 -> f32) ---------------- */

__device__ __forceinline__ v8f wmma16(v16h a, v16h b, v8f c) {
  /* (neg_a, A, neg_b, B, c_mod, C, reuse_a, reuse_b) */
  return __builtin_amdgcn_wmma_f32_16x16x32_f16(false, a, false, b, (short)0, c,
                                                false, false);
}

/* A-fragment (16x32 f16, ISA 7.12.2): lane half=lane/16, row m=lane%16.
   element j<8  -> K = half*8 + j         (contiguous 16B)
   element j>=8 -> K = 16 + half*8 + j-8  (contiguous 16B)                     */
__device__ __forceinline__ v16h frag_a(const _Float16* rowBase, int half) {
  v8h lo = *(const v8h*)(rowBase + half * 8);
  v8h hi = *(const v8h*)(rowBase + 16 + half * 8);
  v16h r;
#pragma unroll
  for (int i = 0; i < 8; ++i) { r[i] = lo[i]; r[i + 8] = hi[i]; }
  return r;
}

/* B-fragment (32x16 f16): lane column n=lane%16, half=lane/16.
   element j<8 -> K = half*8 + j; j>=8 -> K = 16 + half*8 + j-8 (same as A).
   rowBase points at the K-major row for column n (B staged "transposed").    */
__device__ __forceinline__ v16h frag_b(const _Float16* rowBase, int half) {
  v8h lo = *(const v8h*)(rowBase + half * 8);
  v8h hi = *(const v8h*)(rowBase + 16 + half * 8);
  v16h r;
#pragma unroll
  for (int i = 0; i < 8; ++i) { r[i] = lo[i]; r[i + 8] = hi[i]; }
  return r;
}

/* --------------------- TDM 2D tile load (global -> LDS) ------------------- */
#if HAVE_TDM
/* Loads tile_h x tile_w (f16 elements) from a row-major matrix with row
   stride `row_stride` (elements), into LDS at byte offset lds_off with a
   48-half row stride via the D# pad fields (pad_interval=16DW, pad_amount=8DW).
   D# packing per CDNA5 ISA §8.3/8.4; issue from a single wave (EXEC-independent,
   tracked by TENSORcnt). */
__device__ __forceinline__ void tdm_load_2d(const void* gptr, unsigned lds_off,
                                            unsigned row_stride,
                                            unsigned tile_w, unsigned tile_h) {
  unsigned long long ga = (unsigned long long)(uintptr_t)gptr;
  u32x4 g0;
  g0[0] = 1u;                                          /* count=1, user mode   */
  g0[1] = lds_off;                                     /* lds_addr (bytes)     */
  g0[2] = (unsigned)ga;                                /* global_addr[31:0]    */
  g0[3] = (unsigned)((ga >> 32) & 0x01FFFFFFu) | (2u << 30); /* [56:32]|type=2 */
  i32x8 g1;
  g1[0] = (int)((1u << 16) |                           /* data_size = 2 bytes  */
                (1u << 20) |                           /* pad_enable           */
                (3u << 22) |                           /* pad_interval: 16 DW  */
                (7u << 25));                           /* pad_amount:    8 DW  */
  g1[1] = (int)((row_stride & 0xFFFFu) << 16);         /* tensor_dim0[15:0]    */
  g1[2] = (int)(((row_stride >> 16) & 0xFFFFu) |       /* tensor_dim0[31:16]   */
                (0x4000u << 16));                      /* tensor_dim1 = 16384  */
  g1[3] = (int)((tile_w & 0xFFFFu) << 16);             /* dim1 hi=0, tile_dim0 */
  g1[4] = (int)(tile_h & 0xFFFFu);                     /* tile_dim1, tile_dim2=0 */
  g1[5] = (int)row_stride;                             /* dim0_stride[31:0]    */
  g1[6] = 0;                                           /* stride hi, dim1_stride lo */
  g1[7] = 0;
  i32x4 z4 = {};
#if TDM_6ARG
  i32x8 z8 = {};
  __builtin_amdgcn_tensor_load_to_lds(g0, g1, z4, z4, z8, 0);
#else
  __builtin_amdgcn_tensor_load_to_lds(g0, g1, z4, z4, 0);
#endif
}
#endif /* HAVE_TDM */

/* ------------- shared tile engine: C(128x128) += A(MxK) * B(NxK)^T -------- */
/* A row-major [*, lda], B row-major [*, ldb]; both contiguous along K.
   8 waves: wave = wm(0..3)*32 rows x wn(0..1)*64 cols; acc[2][4] 16x16 tiles.
   aLds/bLds are double buffers of 128*LDST halves each.                      */
__device__ __forceinline__ void gemm_bt_accum(
    const _Float16* __restrict__ A, const _Float16* __restrict__ B,
    int lda, int ldb, int K, int m0, int n0,
    _Float16* aLds, _Float16* bLds, v8f acc[2][4])
{
  const int tid  = threadIdx.x;
  const int lane = tid & 31;
  const int wave = tid >> 5;
  const int wm   = wave & 3;
  const int wn   = wave >> 2;
  const int m    = lane & 15;
  const int half = lane >> 4;

#if HAVE_TDM
  /* ---- TDM double-buffered pipeline: wave 0 drives the DMA ---- */
  const unsigned aOff = (unsigned)(uintptr_t)(void*)aLds;  /* LDS byte offset */
  const unsigned bOff = (unsigned)(uintptr_t)(void*)bLds;
  if (wave == 0) {
    tdm_load_2d(A + (size_t)m0 * lda, aOff, (unsigned)lda, 32, 128);
    tdm_load_2d(B + (size_t)n0 * ldb, bOff, (unsigned)ldb, 32, 128);
  }
  for (int kb = 0; kb < K; kb += 32) {
    const int cur = (kb >> 5) & 1;
    if (wave == 0) {
      if (kb + 32 < K) {
        const int nxt = cur ^ 1;
        tdm_load_2d(A + (size_t)m0 * lda + kb + 32,
                    aOff + (unsigned)(nxt * TILEHALVES * 2), (unsigned)lda, 32, 128);
        tdm_load_2d(B + (size_t)n0 * ldb + kb + 32,
                    bOff + (unsigned)(nxt * TILEHALVES * 2), (unsigned)ldb, 32, 128);
        __builtin_amdgcn_s_wait_tensorcnt(2);  /* current tile's 2 DMAs done */
      } else {
        __builtin_amdgcn_s_wait_tensorcnt(0);
      }
    }
    __syncthreads();
    const _Float16* aT = aLds + cur * TILEHALVES;
    const _Float16* bT = bLds + cur * TILEHALVES;
    v16h af[2], bf[4];
#pragma unroll
    for (int mt = 0; mt < 2; ++mt)
      af[mt] = frag_a(aT + (wm * 32 + mt * 16 + m) * LDST, half);
#pragma unroll
    for (int nt = 0; nt < 4; ++nt)
      bf[nt] = frag_b(bT + (wn * 64 + nt * 16 + m) * LDST, half);
#pragma unroll
    for (int mt = 0; mt < 2; ++mt)
#pragma unroll
      for (int nt = 0; nt < 4; ++nt)
        acc[mt][nt] = wmma16(af[mt], bf[nt], acc[mt][nt]);
    __syncthreads();   /* buffer reusable for DMA next iteration */
  }
#else
  /* ---- fallback: synchronous cooperative staging (single buffer) ---- */
  const int ldRow = tid >> 1;
  const int ldCol = (tid & 1) * 16;
  for (int kb = 0; kb < K; kb += 32) {
    *(v16h*)(aLds + ldRow * LDST + ldCol) =
        *(const v16h*)(A + (size_t)(m0 + ldRow) * lda + kb + ldCol);
    *(v16h*)(bLds + ldRow * LDST + ldCol) =
        *(const v16h*)(B + (size_t)(n0 + ldRow) * ldb + kb + ldCol);
    __syncthreads();
    v16h af[2], bf[4];
#pragma unroll
    for (int mt = 0; mt < 2; ++mt)
      af[mt] = frag_a(aLds + (wm * 32 + mt * 16 + m) * LDST, half);
#pragma unroll
    for (int nt = 0; nt < 4; ++nt)
      bf[nt] = frag_b(bLds + (wn * 64 + nt * 16 + m) * LDST, half);
#pragma unroll
    for (int mt = 0; mt < 2; ++mt)
#pragma unroll
      for (int nt = 0; nt < 4; ++nt)
        acc[mt][nt] = wmma16(af[mt], bf[nt], acc[mt][nt]);
    __syncthreads();
  }
#endif
}

/* ------------------------------ kernels ----------------------------------- */

/* vectorized f32 -> f16: 8 elements per thread (two b128 loads, one b128 store) */
__global__ void cvt_f32_f16_v8(const float* __restrict__ src,
                               _Float16* __restrict__ dst, int n8) {
  int i = blockIdx.x * blockDim.x + threadIdx.x;
  if (i < n8) {
    f32x4 lo = *(const f32x4*)(src + (size_t)i * 8);
    f32x4 hi = *(const f32x4*)(src + (size_t)i * 8 + 4);
    v8h o;
#pragma unroll
    for (int j = 0; j < 4; ++j) { o[j] = (_Float16)lo[j]; o[4 + j] = (_Float16)hi[j]; }
    *(v8h*)(dst + (size_t)i * 8) = o;
  }
}

__global__ void zero_f32(float* __restrict__ p, int n) {
  int i = blockIdx.x * blockDim.x + threadIdx.x;
  if (i < n) p[i] = 0.0f;
}

/* QKV GEMM + fused feature map (elu+1 on q,k), f16 outputs */
__global__ void __launch_bounds__(256)
qkv_gemm_kernel(const _Float16* __restrict__ xh, const _Float16* __restrict__ wh,
                _Float16* __restrict__ qh, _Float16* __restrict__ kh,
                _Float16* __restrict__ vh)
{
  __shared__ __align__(128) _Float16 aLds[2 * TILEHALVES];
  __shared__ __align__(128) _Float16 bLds[2 * TILEHALVES];
  const int m0 = blockIdx.y * 128;
  const int n0 = blockIdx.x * 128;            /* 0..3071, 1024-boundaries align */
  v8f acc[2][4] = {};
  gemm_bt_accum(xh, wh, DIM, DIM, DIM, m0, n0, aLds, bLds, acc);

  const int tid = threadIdx.x, lane = tid & 31, wave = tid >> 5;
  const int wm = wave & 3, wn = wave >> 2;
  const int n = lane & 15, half = lane >> 4;
  const int seg = n0 >> 10;                   /* 0=q 1=k 2=v, uniform per block */
  _Float16* dst = (seg == 0) ? qh : (seg == 1) ? kh : vh;
  const int colBase = n0 & 1023;
#pragma unroll
  for (int mt = 0; mt < 2; ++mt) {
    const int gm = m0 + wm * 32 + mt * 16 + half * 8;
#pragma unroll
    for (int nt = 0; nt < 4; ++nt) {
      const int gc = colBase + wn * 64 + nt * 16 + n;
#pragma unroll
      for (int r = 0; r < 8; ++r) {
        float c = acc[mt][nt][r];
        if (seg < 2) c = (c > 0.0f) ? (c + 1.0f) : __expf(c); /* elu(x)+1 */
        dst[(size_t)(gm + r) * DIM + gc] = (_Float16)c;
      }
    }
  }
}

/* per-(b,h,t-chunk): kv[64][64] += k_f^T @ v over 512 tokens; one wave/block */
__global__ void __launch_bounds__(32)
kv_accum_kernel(const _Float16* __restrict__ kh, const _Float16* __restrict__ vh,
                float* __restrict__ kv)
{
  __shared__ _Float16 kT[64 * LDST];   /* [d][t'] */
  __shared__ _Float16 vT[64 * LDST];   /* [e][t'] */
  const int bh = blockIdx.x;                     /* 0..63 */
  const int tc = blockIdx.y;                     /* 0..7  */
  const int b = bh >> 4, h = bh & 15;
  const int lane = threadIdx.x;
  const int m = lane & 15, half = lane >> 4;
  const _Float16* kbase = kh + (size_t)b * SEQ * DIM + h * HD;
  const _Float16* vbase = vh + (size_t)b * SEQ * DIM + h * HD;

  v8f acc[4][4] = {};
  for (int t0 = tc * 512; t0 < tc * 512 + 512; t0 += 32) {
    for (int tp = 0; tp < 32; ++tp) {           /* stage transposed tiles */
      const _Float16* krow = kbase + (size_t)(t0 + tp) * DIM;
      const _Float16* vrow = vbase + (size_t)(t0 + tp) * DIM;
      kT[(lane * 2 + 0) * LDST + tp] = krow[lane * 2 + 0];
      kT[(lane * 2 + 1) * LDST + tp] = krow[lane * 2 + 1];
      vT[(lane * 2 + 0) * LDST + tp] = vrow[lane * 2 + 0];
      vT[(lane * 2 + 1) * LDST + tp] = vrow[lane * 2 + 1];
    }
    __syncthreads();
#pragma unroll
    for (int dt = 0; dt < 4; ++dt) {
      v16h af = frag_a(kT + (dt * 16 + m) * LDST, half);
#pragma unroll
      for (int et = 0; et < 4; ++et) {
        v16h bf = frag_b(vT + (et * 16 + m) * LDST, half);
        acc[dt][et] = wmma16(af, bf, acc[dt][et]);
      }
    }
    __syncthreads();
  }

  float* kvdst = kv + (size_t)bh * (HD * HD);
#pragma unroll
  for (int dt = 0; dt < 4; ++dt)
#pragma unroll
    for (int et = 0; et < 4; ++et)
#pragma unroll
      for (int r = 0; r < 8; ++r) {
        int d = dt * 16 + half * 8 + r;
        int e = et * 16 + m;
        atomicAdd(kvdst + d * HD + e, acc[dt][et][r]);
      }
}

__global__ void __launch_bounds__(64)
ksum_kernel(const _Float16* __restrict__ kh, float* __restrict__ ksum) {
  const int bh = blockIdx.x;
  const int b = bh >> 4, h = bh & 15;
  const int d = threadIdx.x;
  const _Float16* base = kh + (size_t)b * SEQ * DIM + h * HD + d;
  float s = 0.0f;
  for (int t = 0; t < SEQ; ++t) {
    if ((t & 63) == 0 && t + 128 < SEQ)
      __builtin_prefetch(base + (size_t)(t + 128) * DIM, 0, 0); /* global_prefetch_b8 */
    s += (float)base[(size_t)t * DIM];
  }
  ksum[bh * HD + d] = s;
}

/* attn out: q_f @ kv, normalized, f16 into attnh (reuses xh region) */
__global__ void __launch_bounds__(256)
attn_out_kernel(const _Float16* __restrict__ qh, const float* __restrict__ kv,
                const float* __restrict__ ksum, _Float16* __restrict__ attnh)
{
  __shared__ _Float16 kvT[64 * 80];   /* [e][d], stride 80 halves (160B, 32B-mult) */
  __shared__ float normLds[128];
  __shared__ float ksumLds[64];
  const int bh = blockIdx.x;
  const int b = bh >> 4, h = bh & 15;
  const int m0 = blockIdx.y * 128;
  const int tid = threadIdx.x, lane = tid & 31, wave = tid >> 5;
  const int m = lane & 15, half = lane >> 4;

  const float* kvsrc = kv + (size_t)bh * HD * HD;
  for (int i = tid; i < HD * HD; i += 256) {     /* transpose + f32->f16 */
    int d = i >> 6, e = i & 63;
    kvT[e * 80 + d] = (_Float16)kvsrc[i];
  }
  if (tid < 64) ksumLds[tid] = ksum[bh * HD + tid];
  __syncthreads();

  const _Float16* qbase = qh + (size_t)b * SEQ * DIM + h * HD;
  if (tid < 128) {                                /* normalizer per row */
    const _Float16* qrow = qbase + (size_t)(m0 + tid) * DIM;
    float s = 0.0f;
    for (int d = 0; d < HD; ++d) s += (float)qrow[d] * ksumLds[d];
    normLds[tid] = s;
  }
  __syncthreads();

  v8f acc[4] = {};
  const _Float16* arow = qbase + (size_t)(m0 + wave * 16 + m) * DIM;
#pragma unroll
  for (int kb = 0; kb < HD; kb += 32) {
    v16h af = frag_a(arow + kb, half);
#pragma unroll
    for (int nt = 0; nt < 4; ++nt) {
      v16h bf = frag_b(kvT + (nt * 16 + m) * 80 + kb, half);
      acc[nt] = wmma16(af, bf, acc[nt]);
    }
  }

#pragma unroll
  for (int nt = 0; nt < 4; ++nt) {
    int e = nt * 16 + m;
#pragma unroll
    for (int r = 0; r < 8; ++r) {
      int tl = wave * 16 + half * 8 + r;
      float val = acc[nt][r] / (normLds[tl] + 1e-6f);
      attnh[(size_t)(b * SEQ + m0 + tl) * DIM + h * HD + e] = (_Float16)val;
    }
  }
}

/* final projection: attn(f16) @ w_out^T -> fp32 d_out */
__global__ void __launch_bounds__(256)
proj_gemm_kernel(const _Float16* __restrict__ ah, const _Float16* __restrict__ wh,
                 float* __restrict__ out)
{
  __shared__ __align__(128) _Float16 aLds[2 * TILEHALVES];
  __shared__ __align__(128) _Float16 bLds[2 * TILEHALVES];
  const int m0 = blockIdx.y * 128;
  const int n0 = blockIdx.x * 128;
  v8f acc[2][4] = {};
  gemm_bt_accum(ah, wh, DIM, DIM, DIM, m0, n0, aLds, bLds, acc);

  const int tid = threadIdx.x, lane = tid & 31, wave = tid >> 5;
  const int wm = wave & 3, wn = wave >> 2;
  const int n = lane & 15, half = lane >> 4;
#pragma unroll
  for (int mt = 0; mt < 2; ++mt) {
    const int gm = m0 + wm * 32 + mt * 16 + half * 8;
#pragma unroll
    for (int nt = 0; nt < 4; ++nt) {
      const int gn = n0 + wn * 64 + nt * 16 + n;
#pragma unroll
      for (int r = 0; r < 8; ++r)
        out[(size_t)(gm + r) * DIM + gn] = acc[mt][nt][r];
    }
  }
}

/* ------------------------------ launcher ---------------------------------- */

extern "C" void kernel_launch(void* const* d_in, const int* in_sizes, int n_in,
                              void* d_out, int out_size, void* d_ws, size_t ws_size,
                              hipStream_t stream) {
  const float* x    = (const float*)d_in[0];
  const float* wqkv = (const float*)d_in[1];
  const float* wout = (const float*)d_in[2];
  float* out = (float*)d_out;
  char* ws = (char*)d_ws;

  const size_t MB = 1024ull * 1024ull;
  _Float16* xh    = (_Float16*)(ws);                 /* 32MB; reused as attnh  */
  _Float16* wqkvh = (_Float16*)(ws + 32 * MB);       /* 6MB  */
  _Float16* wouth = (_Float16*)(ws + 38 * MB);       /* 2MB  */
  _Float16* qh    = (_Float16*)(ws + 40 * MB);       /* 32MB */
  _Float16* kh    = (_Float16*)(ws + 72 * MB);       /* 32MB */
  _Float16* vh    = (_Float16*)(ws + 104 * MB);      /* 32MB */
  float*    kvbuf = (float*)   (ws + 136 * MB);      /* 1MB  */
  float*    ksum  = (float*)   (ws + 137 * MB);      /* 16KB */

  cvt_f32_f16_v8<<<(MTOT * DIM / 8 + 255) / 256, 256, 0, stream>>>(x, xh, MTOT * DIM / 8);
  cvt_f32_f16_v8<<<(3 * DIM * DIM / 8 + 255) / 256, 256, 0, stream>>>(wqkv, wqkvh, 3 * DIM * DIM / 8);
  cvt_f32_f16_v8<<<(DIM * DIM / 8 + 255) / 256, 256, 0, stream>>>(wout, wouth, DIM * DIM / 8);

  qkv_gemm_kernel<<<dim3(3 * DIM / 128, MTOT / 128), 256, 0, stream>>>(xh, wqkvh, qh, kh, vh);

  zero_f32<<<(BATCH * HEADS * HD * HD + 255) / 256, 256, 0, stream>>>(kvbuf, BATCH * HEADS * HD * HD);
  kv_accum_kernel<<<dim3(BATCH * HEADS, SEQ / 512), 32, 0, stream>>>(kh, vh, kvbuf);
  ksum_kernel<<<BATCH * HEADS, 64, 0, stream>>>(kh, ksum);

  attn_out_kernel<<<dim3(BATCH * HEADS, SEQ / 128), 256, 0, stream>>>(qh, kvbuf, ksum, xh);

  proj_gemm_kernel<<<dim3(DIM / 128, MTOT / 128), 256, 0, stream>>>(xh, wouth, out);
}
